// MambaEEG_23742579212693
// MI455X (gfx1250) — compile-verified, matrix operand order checked by reading, not run on hardware
//
#include <hip/hip_runtime.h>
#include <hip/hip_bf16.h>

typedef __attribute__((ext_vector_type(16))) _Float16 v16h;
typedef __attribute__((ext_vector_type(8)))  float    v8f;

#define BATCH 64
#define C_IN  22
#define SEQ   2048
#define DMODEL 128
#define DI    256
#define DS    16
#define DC    4
#define DR    8
#define NC    4
#define XZW   512           // 2*DI
#define KPAD  32            // C_IN padded to WMMA K
#define NDBC  (DR + 2*DS)   // 40
#define NTILES 32           // XZW / 16
#define BFRAG_ELEMS (NTILES * 32 * 16)   // 16384 halfs = 32 KB

// ---------------------------------------------------------------------------
// Fold emb_w(22x128)@in_w(128x512) directly into per-lane WMMA B-fragment
// layout:  Bfrag[(nt*32 + lane)*16 + j]  where
//   col = nt*16 + (lane&15), kb = (lane<16)?0:8,
//   j<8  -> k = kb+j ;  j>=8 -> k = kb+16+(j-8)   (k>=22 rows are zero)
// Each lane's 16 halfs are contiguous (32 B) -> ds_load_b128 x2 in the GEMM.
// Also b1[n] = emb_b @ in_w[:,n].
// ---------------------------------------------------------------------------
__global__ void MambaEEG_fold_w1(const float* __restrict__ emb_w,
                                 const float* __restrict__ in_w,
                                 const float* __restrict__ emb_b,
                                 _Float16* __restrict__ Bfrag,
                                 float* __restrict__ b1) {
    int idx = blockIdx.x * blockDim.x + threadIdx.x;
    if (idx < BFRAG_ELEMS) {
        int j    = idx & 15;
        int lane = (idx >> 4) & 31;
        int nt   = idx >> 9;
        int col  = nt * 16 + (lane & 15);
        int kb   = (lane < 16) ? 0 : 8;
        int k    = (j < 8) ? (kb + j) : (kb + 16 + (j - 8));
        float acc = 0.f;
        if (k < C_IN)
            for (int d = 0; d < DMODEL; ++d)
                acc += emb_w[k * DMODEL + d] * in_w[d * XZW + col];
        Bfrag[idx] = (_Float16)acc;
    }
    if (idx < XZW) {
        float acc = 0.f;
        for (int d = 0; d < DMODEL; ++d) acc += emb_b[d] * in_w[d * XZW + idx];
        b1[idx] = acc;
    }
}

// Fold out_w(256x128)@fc_w(128x4) -> Wo (256x4)
__global__ void MambaEEG_fold_wo(const float* __restrict__ out_w,
                                 const float* __restrict__ fc_w,
                                 float* __restrict__ Wo) {
    int idx = blockIdx.x * blockDim.x + threadIdx.x;
    if (idx < DI * NC) {
        int d = idx / NC, n = idx % NC;
        float a = 0.f;
        for (int e = 0; e < DMODEL; ++e) a += out_w[d * DMODEL + e] * fc_w[e * NC + n];
        Wo[idx] = a;
    }
}

// ---------------------------------------------------------------------------
// xz[(b*L+l), 0:512] = X[(b,l), 0:22] @ W1 + b1     (WMMA f16, K = 32 padded)
// 8 waves/block; each wave owns one 16-row M tile, loops 32 N tiles.
// Pre-swizzled B fragments staged in LDS via GLOBAL_LOAD_ASYNC_TO_LDS_B128
// (ASYNCcnt path, no VGPR round-trip); each lane then pulls its v16h with
// two b128 LDS reads.
// ---------------------------------------------------------------------------
__global__ __launch_bounds__(256) void MambaEEG_inproj_wmma(
    const float* __restrict__ x, const _Float16* __restrict__ Bfrag,
    const float* __restrict__ b1, _Float16* __restrict__ xz) {
    __shared__ _Float16 sB[BFRAG_ELEMS];     // 32 KB, fragment layout
    {
        // Async bulk copy global -> LDS: 256 lanes x 16 B x 8 issues = 32 KB.
        // Low 32 bits of the generic LDS pointer are the LDS byte offset
        // (flat-aperture rule), which is what VDST carries for async loads.
        unsigned lbase = (unsigned)(size_t)(&sB[0]) + (unsigned)(threadIdx.x * 16);
        unsigned long long gbase =
            (unsigned long long)(size_t)Bfrag + (unsigned long long)(threadIdx.x * 16);
#pragma unroll
        for (int it = 0; it < 8; ++it) {
            unsigned loff = lbase + (unsigned)(it * 4096);
            unsigned long long ga = gbase + (unsigned long long)(it * 4096);
            asm volatile("global_load_async_to_lds_b128 %0, %1, off"
                         :: "v"(loff), "v"(ga) : "memory");
        }
        asm volatile("s_wait_asynccnt 0x0" ::: "memory");
    }
    __syncthreads();

    const int wave = threadIdx.x >> 5;
    const int lane = threadIdx.x & 31;
    const int mtile = blockIdx.x * 8 + wave;     // 8192 M tiles total
    const int b = mtile >> 7;                    // 128 tiles per batch row
    const int l = ((mtile & 127) << 4) + (lane & 15);
    const int kb = (lane < 16) ? 0 : 8;          // half-wave K split (A side)

    // A fragment: 16x32 f16 (rows = (b,l), cols = padded channel)
    const float* xb = x + (size_t)b * C_IN * SEQ + l;
    v16h a;
#pragma unroll
    for (int j = 0; j < 8; ++j) {
        int c = kb + j;
        a[j] = (c < C_IN) ? (_Float16)xb[(size_t)c * SEQ] : (_Float16)0.f;
    }
#pragma unroll
    for (int j = 0; j < 8; ++j) {
        int c = kb + 16 + j;
        a[8 + j] = (c < C_IN) ? (_Float16)xb[(size_t)c * SEQ] : (_Float16)0.f;
    }

    const size_t mrowbase = (size_t)mtile * 16;
    const int mbase = (lane < 16) ? 0 : 8;
    const _Float16* myfrag = &sB[lane * 16];
    for (int nt = 0; nt < NTILES; ++nt) {
        const int col = nt * 16 + (lane & 15);
        v16h bf = *(const v16h*)(myfrag + nt * (32 * 16));  // 2x ds_load_b128

        v8f cacc = {};
        cacc = __builtin_amdgcn_wmma_f32_16x16x32_f16(
            false, a, false, bf, (short)0, cacc, false, false);

        const float bias = b1[col];
#pragma unroll
        for (int i = 0; i < 8; ++i) {
            size_t m = mrowbase + mbase + i;
            xz[m * XZW + col] = (_Float16)(cacc[i] + bias);
        }
    }
}

// ---------------------------------------------------------------------------
// Fused: depthwise conv4 + SiLU -> xproj(256->40) -> dt proj + softplus ->
// selective scan (16 states in regs) -> D skip + SiLU(z) gate -> mean pool.
// One block per batch element; thread d owns channel d. xproj_w staged in
// LDS transposed ([o][i], stride-1 per lane -> bank-conflict-free); dt_w
// column hoisted to registers.
// ---------------------------------------------------------------------------
__global__ __launch_bounds__(DI) void MambaEEG_fused_scan(
    const _Float16* __restrict__ xz, const float* __restrict__ conv_w,
    const float* __restrict__ conv_b, const float* __restrict__ xproj_w,
    const float* __restrict__ dt_w, const float* __restrict__ dt_b,
    const float* __restrict__ A_log, const float* __restrict__ D_skip,
    float* __restrict__ ybar) {
    __shared__ float s_xp[NDBC * DI];    // 40 KB, transposed xproj_w
    __shared__ float s_xc[DI];
    __shared__ float s_dbc[NDBC];

    const int b = blockIdx.x;
    const int d = threadIdx.x;
    const int wave = d >> 5, lane = d & 31;

    for (int idx = d; idx < NDBC * DI; idx += DI) {
        int o = idx >> 8, i = idx & 255;          // DI == 256
        s_xp[idx] = xproj_w[i * NDBC + o];
    }

    const float cw0 = conv_w[d * DC + 0], cw1 = conv_w[d * DC + 1];
    const float cw2 = conv_w[d * DC + 2], cw3 = conv_w[d * DC + 3];
    const float cb = conv_b[d];
    const float dtb = dt_b[d];
    const float dsk = D_skip[d];

    float dtw[DR];
#pragma unroll
    for (int r = 0; r < DR; ++r) dtw[r] = dt_w[r * DI + d];

    float Ad[DS];
#pragma unroll
    for (int s = 0; s < DS; ++s) Ad[s] = -__expf(A_log[d * DS + s]);

    float st[DS];
#pragma unroll
    for (int s = 0; s < DS; ++s) st[s] = 0.f;

    float h1 = 0.f, h2 = 0.f, h3 = 0.f;   // xm at l-1, l-2, l-3 (causal pad)
    float acc = 0.f;

    __syncthreads();

    const _Float16* xzb = xz + (size_t)b * SEQ * XZW;
    for (int l = 0; l < SEQ; ++l) {
        const _Float16* row = xzb + (size_t)l * XZW;
        const float xm = (float)row[d];
        const float z  = (float)row[DI + d];
        if (l + 16 < SEQ) __builtin_prefetch(row + 16 * XZW + d, 0, 1);

        float xc = cw3 * xm + cw2 * h1 + cw1 * h2 + cw0 * h3 + cb;
        xc = xc / (1.f + __expf(-xc));            // SiLU
        h3 = h2; h2 = h1; h1 = xm;

        s_xc[d] = xc;
        __syncthreads();

        // dbc[0:40] = xc @ xproj_w ; 8 waves x wave32 shuffle reduction
        for (int o = wave; o < NDBC; o += 8) {
            const float* xp = &s_xp[o * DI];
            float p = 0.f;
            for (int i = lane; i < DI; i += 32) p += s_xc[i] * xp[i];
#pragma unroll
            for (int off = 16; off >= 1; off >>= 1) p += __shfl_xor(p, off, 32);
            if (lane == 0) s_dbc[o] = p;
        }
        __syncthreads();

        float dtacc = dtb;
#pragma unroll
        for (int r = 0; r < DR; ++r) dtacc += s_dbc[r] * dtw[r];
        const float dt = (dtacc > 20.f) ? dtacc : __logf(1.f + __expf(dtacc));

        const float dx = dt * xc;
        float y = 0.f;
#pragma unroll
        for (int s = 0; s < DS; ++s) {
            const float dA = __expf(dt * Ad[s]);
            st[s] = st[s] * dA + dx * s_dbc[DR + s];        // B_t
            y += st[s] * s_dbc[DR + DS + s];                // C_t
        }
        const float zs = z / (1.f + __expf(-z));
        acc += (y + dsk * xc) * zs;
    }
    ybar[b * DI + d] = acc * (1.f / (float)SEQ);
}

// logits[b,n] = ybar[b,:] @ Wo[:,n] + fc_b[n]
__global__ void MambaEEG_final(const float* __restrict__ ybar,
                               const float* __restrict__ Wo,
                               const float* __restrict__ fc_b,
                               float* __restrict__ out) {
    int idx = blockIdx.x * blockDim.x + threadIdx.x;
    if (idx < BATCH * NC) {
        int b = idx / NC, n = idx % NC;
        float a = fc_b[n];
        for (int d = 0; d < DI; ++d) a += ybar[b * DI + d] * Wo[d * NC + n];
        out[idx] = a;
    }
}

extern "C" void kernel_launch(void* const* d_in, const int* in_sizes, int n_in,
                              void* d_out, int out_size, void* d_ws, size_t ws_size,
                              hipStream_t stream) {
    const float* x       = (const float*)d_in[0];
    const float* emb_w   = (const float*)d_in[1];
    const float* emb_b   = (const float*)d_in[2];
    const float* in_w    = (const float*)d_in[3];
    const float* conv_w  = (const float*)d_in[4];
    const float* conv_b  = (const float*)d_in[5];
    const float* xproj_w = (const float*)d_in[6];
    const float* dt_w    = (const float*)d_in[7];
    const float* dt_b    = (const float*)d_in[8];
    const float* A_log   = (const float*)d_in[9];
    const float* D_skip  = (const float*)d_in[10];
    const float* out_w   = (const float*)d_in[11];
    const float* fc_w    = (const float*)d_in[12];
    const float* fc_b    = (const float*)d_in[13];

    // workspace layout
    char* ws = (char*)d_ws;
    _Float16* xzf16 = (_Float16*)ws;                       // 64*2048*512 f16 = 128 MB
    size_t off = (size_t)BATCH * SEQ * XZW * sizeof(_Float16);
    _Float16* Bfrag = (_Float16*)(ws + off); off += (size_t)BFRAG_ELEMS * sizeof(_Float16);
    float* b1    = (float*)(ws + off);    off += (size_t)XZW * sizeof(float);
    float* Wo    = (float*)(ws + off);    off += (size_t)DI * NC * sizeof(float);
    float* ybar  = (float*)(ws + off);    off += (size_t)BATCH * DI * sizeof(float);

    MambaEEG_fold_w1<<<(BFRAG_ELEMS + 255) / 256, 256, 0, stream>>>(emb_w, in_w, emb_b, Bfrag, b1);
    MambaEEG_fold_wo<<<(DI * NC + 255) / 256, 256, 0, stream>>>(out_w, fc_w, Wo);
    MambaEEG_inproj_wmma<<<1024, 256, 0, stream>>>(x, Bfrag, b1, xzf16);
    MambaEEG_fused_scan<<<BATCH, DI, 0, stream>>>(xzf16, conv_w, conv_b, xproj_w,
                                                  dt_w, dt_b, A_log, D_skip, ybar);
    MambaEEG_final<<<1, 256, 0, stream>>>(ybar, Wo, fc_b, (float*)d_out);
}
